// MARONetworkTeacherForcing_30391188587243
// MI455X (gfx1250) — matrix operand order repaired and checked by reading, not compiled, not run on hardware
//
#include <hip/hip_runtime.h>
#include <hip/hip_bf16.h>

// ---------------------------------------------------------------------------
// Persistent fused LSTM-rollout for MI455X (gfx1250, wave32, WMMA).
//   BS=128, T=512 (511 steps), NA=8, OD=32, HID=512, IN=256, GATES=2048.
// Grid: 32 WGs x 256 threads (8 waves). Weights f16 in LDS (staged once),
// recurrent vectors (x, h, mixed) as f16 in workspace / L2, c + loss in VGPRs.
// Matmuls: v_wmma_f32_16x16x32_f16. Grid barrier 3x per step via agent atomics.
// ---------------------------------------------------------------------------

typedef _Float16 v16h __attribute__((ext_vector_type(16)));
typedef _Float16 v8h  __attribute__((ext_vector_type(8)));
typedef float    v8f  __attribute__((ext_vector_type(8)));

#define BS      128
#define TT      512
#define TSTEPS  511
#define HID     512
#define INDIM   256
#define NWG     32
#define HALF_LOG_2PI 0.91893853320467274178f

// ---- workspace layout (bytes, all 256-aligned) ----
#define OFF_BAR   0          // 2 x u32 (counter @ +0, generation @ +64)
#define OFF_LOSS  256        // 128 f32 per-batch masked logprob sums
#define OFF_W1    1024       // 512*256  f16
#define OFF_WIH   263168     // 2048*512 f16
#define OFF_WHH   2360320    // 2048*512 f16
#define OFF_WOBS  4457472    // 512*512  f16
#define OFF_X     4981760    // 128*512  f16
#define OFF_H0    5112832    // 128*512  f16
#define OFF_H1    5243904    // 128*512  f16
#define OFF_MIX   5374976    // 128*256  f16
#define WS_NEED   5440512

// ----------------------------------------------------------------------------
// WMMA helpers: wave32 fragment loaders matching CDNA5 ISA 7.12.2 layouts.
// ----------------------------------------------------------------------------
__device__ inline v8f wmma_f16(v16h a, v16h b, v8f c) {
  return __builtin_amdgcn_wmma_f32_16x16x32_f16(false, a, false, b,
                                                (short)0, c, false, false);
}

// A (16x32 f16, row-major, leading dim `ld` halves). Per ISA: lane L<16 holds
// row L, K = {0..7, 16..23}; lane L>=16 holds row L-16, K = {8..15, 24..31}.
__device__ inline v16h load_a_frag(const _Float16* tile, int ld) {
  const int lane = threadIdx.x & 31;
  const int row  = lane & 15;
  const int off  = (lane & 16) ? 8 : 0;
  const _Float16* p = tile + row * ld + off;
  v8h lo = *(const v8h*)(p);        // K = off .. off+7
  v8h hi = *(const v8h*)(p + 16);   // K = 16+off .. 23+off
  v16h a;
#pragma unroll
  for (int i = 0; i < 8; ++i) { a[i] = lo[i]; a[i + 8] = hi[i]; }
  return a;
}

// B (32x16 f16) with B(k,n) = W[n][k] (we compute X @ W^T): lane L supplies
// column n = colbase + (L&15), K = kbase + (L<16 ? {0..7,16..23} : {8..15,24..31})
// — two 8-half reads from row n of W (leading dim ldk halves).
__device__ inline v16h load_b_frag(const _Float16* w, int ldk, int colbase, int kbase) {
  const int lane = threadIdx.x & 31;
  const int col  = colbase + (lane & 15);
  const int ko   = kbase + ((lane & 16) ? 8 : 0);
  const _Float16* p = w + col * ldk + ko;
  v8h lo = *(const v8h*)(p);
  v8h hi = *(const v8h*)(p + 16);
  v16h b;
#pragma unroll
  for (int i = 0; i < 8; ++i) { b[i] = lo[i]; b[i + 8] = hi[i]; }
  return b;
}

__device__ inline float sigm(float x) { return 1.0f / (1.0f + __expf(-x)); }

// ----------------------------------------------------------------------------
// Grid-wide barrier (persistent kernel, all 32 WGs resident).
// ----------------------------------------------------------------------------
__device__ inline void grid_barrier(unsigned* bar) {
  __syncthreads();
  if (threadIdx.x == 0) {
    __threadfence();                      // make phase writes visible (agent)
    unsigned* cnt = bar;
    unsigned* gen = bar + 16;             // 64B apart
    unsigned g = __hip_atomic_load(gen, __ATOMIC_RELAXED, __HIP_MEMORY_SCOPE_AGENT);
    unsigned a = __hip_atomic_fetch_add(cnt, 1u, __ATOMIC_ACQ_REL, __HIP_MEMORY_SCOPE_AGENT);
    if (a == (unsigned)NWG - 1u) {
      __hip_atomic_store(cnt, 0u, __ATOMIC_RELAXED, __HIP_MEMORY_SCOPE_AGENT);
      __hip_atomic_store(gen, g + 1u, __ATOMIC_RELEASE, __HIP_MEMORY_SCOPE_AGENT);
    } else {
      while (__hip_atomic_load(gen, __ATOMIC_ACQUIRE, __HIP_MEMORY_SCOPE_AGENT) == g)
        __builtin_amdgcn_s_sleep(2);
    }
  }
  __syncthreads();
}

// ----------------------------------------------------------------------------
// Prep kernels
// ----------------------------------------------------------------------------
__global__ void k_f32_to_f16(const float* __restrict__ src,
                             _Float16* __restrict__ dst, int n) {
  int i = blockIdx.x * blockDim.x + threadIdx.x;
  if (i < n) dst[i] = (_Float16)src[i];
}

__global__ void k_init(float* __restrict__ loss_num, unsigned* __restrict__ bar,
                       _Float16* __restrict__ h0, _Float16* __restrict__ h1,
                       _Float16* __restrict__ mix, const float* __restrict__ data) {
  int i = blockIdx.x * blockDim.x + threadIdx.x;        // 65536 threads
  if (i < 32) bar[i] = 0u;
  if (i < BS) loss_num[i] = 0.0f;
  if (i < BS * HID) { h0[i] = (_Float16)0.0f; h1[i] = (_Float16)0.0f; }
  if (i < BS * INDIM) {
    int b = i >> 8, col = i & 255;                      // step 0: keep==true
    mix[i] = (_Float16)data[((size_t)b * TT + 0) * INDIM + col];
  }
}

// ----------------------------------------------------------------------------
// The persistent rollout kernel.
// ----------------------------------------------------------------------------
__global__ __launch_bounds__(256, 1)
void k_persist(const float* __restrict__ data, const float* __restrict__ mask,
               const unsigned char* __restrict__ keep,
               const float* __restrict__ b1, const float* __restrict__ bih,
               const float* __restrict__ bhh, const float* __restrict__ bobs,
               const _Float16* __restrict__ gW1, const _Float16* __restrict__ gWih,
               const _Float16* __restrict__ gWhh, const _Float16* __restrict__ gWo,
               _Float16* __restrict__ xbuf, _Float16* __restrict__ h0,
               _Float16* __restrict__ h1, _Float16* __restrict__ mixbuf,
               float* __restrict__ loss_num, unsigned* __restrict__ bar) {
  extern __shared__ _Float16 smem[];
  _Float16* sWih = smem;                  // 64 rows x 512 (i,f,g,o x 16 cols)
  _Float16* sWhh = sWih + 64 * 512;       // 64 rows x 512
  _Float16* sW1  = sWhh + 64 * 512;       // 16 rows x 256
  _Float16* sWo  = sW1 + 16 * 256;        // 32 rows x 512 (mu cols, logsig cols)

  const int wg   = blockIdx.x;            // 0..31
  const int tid  = threadIdx.x;
  const int wave = tid >> 5;              // 0..7  -> m-block for every phase
  const int lane = tid & 31;
  const int lc   = lane & 15;
  const int hi8  = (lane & 16) ? 8 : 0;
  const int m0   = wave * 16;

  // ---- stage weight slices into LDS (once; reused for all 511 steps) ----
  for (int idx = tid; idx < 64 * 64; idx += 256) {
    int r = idx >> 6, c8 = (idx & 63) << 3;
    int grow = (r >> 4) * 512 + wg * 16 + (r & 15);     // gate g, col jb*16+c
    *(v8h*)(sWih + r * 512 + c8) = *(const v8h*)(gWih + (size_t)grow * 512 + c8);
    *(v8h*)(sWhh + r * 512 + c8) = *(const v8h*)(gWhh + (size_t)grow * 512 + c8);
  }
  for (int idx = tid; idx < 16 * 32; idx += 256) {
    int r = idx >> 5, c8 = (idx & 31) << 3;
    *(v8h*)(sW1 + r * 256 + c8) = *(const v8h*)(gW1 + (size_t)(wg * 16 + r) * 256 + c8);
  }
  if (wg < 16) {
    for (int idx = tid; idx < 32 * 64; idx += 256) {
      int r = idx >> 6, c8 = (idx & 63) << 3;
      int grow = (r < 16) ? (wg * 16 + r) : (256 + wg * 16 + (r - 16));
      *(v8h*)(sWo + r * 512 + c8) = *(const v8h*)(gWo + (size_t)grow * 512 + c8);
    }
  }
  __syncthreads();

  // ---- persistent per-lane state ----
  float creg[8], lsum[8];
#pragma unroll
  for (int r = 0; r < 8; ++r) { creg[r] = 0.0f; lsum[r] = 0.0f; }

  const float b1v = b1[wg * 16 + lc];
  float bg[4];
#pragma unroll
  for (int g = 0; g < 4; ++g)
    bg[g] = bih[g * 512 + wg * 16 + lc] + bhh[g * 512 + wg * 16 + lc];
  float bmu = 0.0f, bls = 0.0f;
  if (wg < 16) { bmu = bobs[wg * 16 + lc]; bls = bobs[256 + wg * 16 + lc]; }

  for (int t = 0; t < TSTEPS; ++t) {
    const _Float16* hcur  = (t & 1) ? h1 : h0;
    _Float16*       hnext = (t & 1) ? h0 : h1;

    // ---- Phase A: x = relu(mixed @ W1^T + b1), cols [wg*16, wg*16+16) ----
    {
      v8f acc = {};
#pragma unroll
      for (int kc = 0; kc < 8; ++kc) {
        v16h a = load_a_frag(mixbuf + m0 * INDIM + kc * 32, INDIM);
        acc = wmma_f16(a, load_b_frag(sW1, 256, 0, kc * 32), acc);
      }
#pragma unroll
      for (int r = 0; r < 8; ++r) {
        float v = acc[r] + b1v;
        v = v > 0.0f ? v : 0.0f;
        xbuf[(m0 + r + hi8) * HID + wg * 16 + lc] = (_Float16)v;
      }
    }
    grid_barrier(bar);

    // ---- Phase B: gates + LSTM cell for hidden cols [wg*16, wg*16+16) ----
    {
      v8f a0 = {}, a1 = {}, a2 = {}, a3 = {};
#pragma unroll 4
      for (int kc = 0; kc < 16; ++kc) {
        v16h a = load_a_frag(xbuf + m0 * HID + kc * 32, HID);
        a0 = wmma_f16(a, load_b_frag(sWih, 512, 0,  kc * 32), a0);
        a1 = wmma_f16(a, load_b_frag(sWih, 512, 16, kc * 32), a1);
        a2 = wmma_f16(a, load_b_frag(sWih, 512, 32, kc * 32), a2);
        a3 = wmma_f16(a, load_b_frag(sWih, 512, 48, kc * 32), a3);
      }
#pragma unroll 4
      for (int kc = 0; kc < 16; ++kc) {
        v16h a = load_a_frag(hcur + m0 * HID + kc * 32, HID);
        a0 = wmma_f16(a, load_b_frag(sWhh, 512, 0,  kc * 32), a0);
        a1 = wmma_f16(a, load_b_frag(sWhh, 512, 16, kc * 32), a1);
        a2 = wmma_f16(a, load_b_frag(sWhh, 512, 32, kc * 32), a2);
        a3 = wmma_f16(a, load_b_frag(sWhh, 512, 48, kc * 32), a3);
      }
#pragma unroll
      for (int r = 0; r < 8; ++r) {
        float gi = sigm(a0[r] + bg[0]);
        float gf = sigm(a1[r] + bg[1]);
        float gg = tanhf(a2[r] + bg[2]);
        float go = sigm(a3[r] + bg[3]);
        float c  = gf * creg[r] + gi * gg;
        creg[r]  = c;                                // cell state lives in VGPRs
        float h  = go * tanhf(c);
        hnext[(m0 + r + hi8) * HID + wg * 16 + lc] = (_Float16)h;
      }
    }
    grid_barrier(bar);

    // ---- Phase C (wg<16): out = h @ Wobs^T, loss, next-step mixed ----
    if (wg < 16) {
      v8f amu = {}, als = {};
#pragma unroll 4
      for (int kc = 0; kc < 16; ++kc) {
        v16h a = load_a_frag(hnext + m0 * HID + kc * 32, HID);
        amu = wmma_f16(a, load_b_frag(sWo, 512, 0,  kc * 32), amu);
        als = wmma_f16(a, load_b_frag(sWo, 512, 16, kc * 32), als);
      }
      const int col = wg * 16 + lc;                  // obs column 0..255
      const int na  = col >> 5;
#pragma unroll
      for (int r = 0; r < 8; ++r) {
        int row = m0 + r + hi8;                      // batch index
        float mu = amu[r] + bmu;
        float ls = als[r] + bls;                     // log sigma
        const float* dptr = data + ((size_t)row * TT + t) * INDIM + col;
        float d0 = dptr[0];
        float d1 = dptr[INDIM];                      // data[row][t+1][col]
        float inv_sig = __expf(-ls);
        float z  = (d1 - d0 - mu) * inv_sig;
        float lp = -0.5f * z * z - ls - HALF_LOG_2PI;
        lsum[r] += mask[(size_t)row * TT + t] * lp;
        if (t < TSTEPS - 1) {                        // mixed for step t+1
          unsigned char k = keep[((size_t)(t + 1) * BS + row) * 8 + na];
          float mv = k ? d1 : (d0 + mu);             // last_pred = d_t + mu
          mixbuf[row * INDIM + col] = (_Float16)mv;
        }
      }
    }
    grid_barrier(bar);
  }

  if (wg < 16) {
#pragma unroll
    for (int r = 0; r < 8; ++r)
      atomicAdd(&loss_num[m0 + r + hi8], lsum[r]);
  }
}

// ----------------------------------------------------------------------------
// Finalize: obs_loss_b = -num_b / (NA * sum_t mask[b,t]);  out = mean_b.
// ----------------------------------------------------------------------------
__global__ void k_final(const float* __restrict__ loss_num,
                        const float* __restrict__ mask, float* __restrict__ out) {
  __shared__ float red[BS];
  int b = threadIdx.x;                               // 128 threads
  float den = 0.0f;
  for (int t = 0; t < TSTEPS; ++t) den += mask[(size_t)b * TT + t];
  red[b] = -loss_num[b] / (den * 8.0f);
  __syncthreads();
  for (int s = 64; s > 0; s >>= 1) {
    if (b < s) red[b] += red[b + s];
    __syncthreads();
  }
  if (b == 0) out[0] = red[0] / (float)BS;
}

// ----------------------------------------------------------------------------
extern "C" void kernel_launch(void* const* d_in, const int* in_sizes, int n_in,
                              void* d_out, int out_size, void* d_ws, size_t ws_size,
                              hipStream_t stream) {
  const float*         data = (const float*)d_in[0];
  const float*         mask = (const float*)d_in[1];
  const unsigned char* keep = (const unsigned char*)d_in[2];  // jax bool -> u8
  const float* W1   = (const float*)d_in[3];
  const float* b1   = (const float*)d_in[4];
  const float* Wih  = (const float*)d_in[5];
  const float* Whh  = (const float*)d_in[6];
  const float* bih  = (const float*)d_in[7];
  const float* bhh  = (const float*)d_in[8];
  const float* Wobs = (const float*)d_in[9];
  const float* bobs = (const float*)d_in[10];
  (void)in_sizes; (void)n_in; (void)out_size; (void)ws_size;

  char* ws = (char*)d_ws;
  unsigned* bar   = (unsigned*)(ws + OFF_BAR);
  float*    loss  = (float*)(ws + OFF_LOSS);
  _Float16* w1h   = (_Float16*)(ws + OFF_W1);
  _Float16* wihh  = (_Float16*)(ws + OFF_WIH);
  _Float16* whhh  = (_Float16*)(ws + OFF_WHH);
  _Float16* wobsh = (_Float16*)(ws + OFF_WOBS);
  _Float16* xbuf  = (_Float16*)(ws + OFF_X);
  _Float16* h0    = (_Float16*)(ws + OFF_H0);
  _Float16* h1    = (_Float16*)(ws + OFF_H1);
  _Float16* mixb  = (_Float16*)(ws + OFF_MIX);

  // f32 -> f16 weight conversion (idempotent, deterministic)
  k_f32_to_f16<<<(512 * 256 + 255) / 256, 256, 0, stream>>>(W1, w1h, 512 * 256);
  k_f32_to_f16<<<(2048 * 512 + 255) / 256, 256, 0, stream>>>(Wih, wihh, 2048 * 512);
  k_f32_to_f16<<<(2048 * 512 + 255) / 256, 256, 0, stream>>>(Whh, whhh, 2048 * 512);
  k_f32_to_f16<<<(512 * 512 + 255) / 256, 256, 0, stream>>>(Wobs, wobsh, 512 * 512);

  // barrier / loss / h / mixed(t=0) init
  k_init<<<256, 256, 0, stream>>>(loss, bar, h0, h1, mixb, data);

  // persistent rollout: 32 WGs x 256 threads, 172 KB dynamic LDS per WG
  // (gfx1250 WGP has 320 KB LDS -> 1 WG per WGP, 32 WGPs co-resident).
  const size_t smem = (size_t)(64 * 512 * 2 + 16 * 256 + 32 * 512) * sizeof(_Float16);
  k_persist<<<dim3(NWG), dim3(256), smem, stream>>>(
      data, mask, keep, b1, bih, bhh, bobs,
      w1h, wihh, whhh, wobsh, xbuf, h0, h1, mixb, loss, bar);

  k_final<<<1, BS, 0, stream>>>(loss, mask, (float*)d_out);
}
